// GraphTransformer_31894427140810
// MI455X (gfx1250) — matrix operand order repaired and checked by reading, not compiled, not run on hardware
//
#include <hip/hip_runtime.h>
#include <hip/hip_bf16.h>
#include <stdint.h>

typedef __attribute__((ext_vector_type(16))) _Float16 v16h;
typedef __attribute__((ext_vector_type(8)))  float    v8f;

#define N_NODES 16384
#define NB      64
#define NPG     256
#define NE      524288
#define SQ      257            // 1 cls + 256 nodes
#define ROWS    (NB * SQ)      // 16448 (= 1028 * 16)
#define WDIM    256
#define DFFD    1024
#define HEADS   8
#define DH      32
#define HIDD    512
#define OUTD    64

__device__ __forceinline__ float gelu_f(float x) {
    const float c = 0.7978845608028654f;  // sqrt(2/pi)
    return 0.5f * x * (1.0f + tanhf(c * (x + 0.044715f * x * x * x)));
}

// ---------------------------------------------------------------------------
// 1. Positional encoding + concat + GCN feature transform: xw = [x | pe] @ gcn_w
// ---------------------------------------------------------------------------
__global__ void k_pe_gcn_xw(const float* __restrict__ x, const float* __restrict__ p,
                            const float* __restrict__ gw, float* __restrict__ xw) {
    __shared__ float xc[56];
    const int row = blockIdx.x, t = threadIdx.x;
    if (t < 32) {
        xc[t] = x[row * 32 + t];
    } else if (t < 56) {
        const int idx = t - 32, d = idx >> 3, i = idx & 7;
        xc[t] = sinf(p[row * 3 + d] * 3.14159265358979323846f * (float)(1 << i));
    }
    __syncthreads();
    float acc = 0.0f;
    for (int k = 0; k < 56; ++k) acc += xc[k] * gw[k * WDIM + t];
    xw[(size_t)row * WDIM + t] = acc;
}

// ---------------------------------------------------------------------------
// 2. Degrees (with self loop), dinv = rsqrt(max(deg,1e-12))
// ---------------------------------------------------------------------------
__global__ void k_fill_deg(float* __restrict__ deg) {
    deg[blockIdx.x * 256 + threadIdx.x] = 1.0f;  // self loop
}
__global__ void k_deg_edges(const int* __restrict__ edge, float* __restrict__ deg) {
    const int e = blockIdx.x * 256 + threadIdx.x;
    atomicAdd(&deg[edge[NE + e]], 1.0f);         // segment over col
}
__global__ void k_dinv(const float* __restrict__ deg, float* __restrict__ dinv) {
    const int i = blockIdx.x * 256 + threadIdx.x;
    dinv[i] = rsqrtf(fmaxf(deg[i], 1e-12f));
}

// 3. h init: self-loop term + bias;  4. edge scatter: h[col] += dinv[r]*dinv[c]*xw[row]
__global__ void k_h_init(const float* __restrict__ xw, const float* __restrict__ dinv,
                         const float* __restrict__ gcn_b, float* __restrict__ h) {
    const int row = blockIdx.x, t = threadIdx.x;
    const float di = dinv[row];
    h[(size_t)row * WDIM + t] = di * di * xw[(size_t)row * WDIM + t] + gcn_b[t];
}
__global__ void k_h_edges(const int* __restrict__ edge, const float* __restrict__ dinv,
                          const float* __restrict__ xw, float* __restrict__ h) {
    const int gid = blockIdx.x * 256 + threadIdx.x;
    const int e = gid >> 6, c4 = (gid & 63) << 2;
    const int r = edge[e], c = edge[NE + e];
    const float cf = dinv[r] * dinv[c];
    const float4 v = *(const float4*)(xw + (size_t)r * WDIM + c4);
    float* dst = h + (size_t)c * WDIM + c4;
    atomicAdd(dst + 0, cf * v.x);
    atomicAdd(dst + 1, cf * v.y);
    atomicAdd(dst + 2, cf * v.z);
    atomicAdd(dst + 3, cf * v.w);
}

// ---------------------------------------------------------------------------
// 5. Build token matrix (cls + dense reshape) fused with ln_pre -> xd (f32)
// ---------------------------------------------------------------------------
__global__ void k_build_xd(const float* __restrict__ h, const float* __restrict__ cls,
                           const float* __restrict__ s, const float* __restrict__ b,
                           float* __restrict__ xd) {
    __shared__ float red[256];
    const int r = blockIdx.x, t = threadIdx.x;
    const int bb = r / SQ, sI = r - bb * SQ;
    float v = (sI == 0) ? cls[t] : h[(size_t)(bb * NPG + sI - 1) * WDIM + t];
    red[t] = v; __syncthreads();
    for (int st = 128; st > 0; st >>= 1) { if (t < st) red[t] += red[t + st]; __syncthreads(); }
    const float mean = red[0] * (1.0f / 256.0f); __syncthreads();
    const float d = v - mean;
    red[t] = d * d; __syncthreads();
    for (int st = 128; st > 0; st >>= 1) { if (t < st) red[t] += red[t + st]; __syncthreads(); }
    const float var = red[0] * (1.0f / 256.0f);
    xd[(size_t)r * WDIM + t] = d * rsqrtf(var + 1e-5f) * s[t] + b[t];
}

// Generic LayerNorm over 256 dims, f32 in -> f16 out (GEMM A operand)
__global__ void k_layernorm_f16(const float* __restrict__ in, const float* __restrict__ s,
                                const float* __restrict__ b, _Float16* __restrict__ out) {
    __shared__ float red[256];
    const int r = blockIdx.x, t = threadIdx.x;
    const float v = in[(size_t)r * WDIM + t];
    red[t] = v; __syncthreads();
    for (int st = 128; st > 0; st >>= 1) { if (t < st) red[t] += red[t + st]; __syncthreads(); }
    const float mean = red[0] * (1.0f / 256.0f); __syncthreads();
    const float d = v - mean;
    red[t] = d * d; __syncthreads();
    for (int st = 128; st > 0; st >>= 1) { if (t < st) red[t] += red[t + st]; __syncthreads(); }
    const float var = red[0] * (1.0f / 256.0f);
    out[(size_t)r * WDIM + t] = (_Float16)(d * rsqrtf(var + 1e-5f) * s[t] + b[t]);
}

__global__ void k_f32_f16(const float* __restrict__ src, _Float16* __restrict__ dst, int n) {
    const int i = blockIdx.x * 256 + threadIdx.x;
    if (i < n) dst[i] = (_Float16)src[i];
}

// ---------------------------------------------------------------------------
// 6. f16 WMMA GEMM: out = act(A @ W + bias) [+ res]
//    Block = 8 waves; tile 128x64; wave = 16 rows x 64 cols (4 accumulators).
//    Double-buffered LDS (ping-pong): store staged regs -> barrier -> issue next
//    tile's global loads -> WMMA on current buffer. One barrier per k-step.
//    Loader roles: threads 0-127 stage A rows (4x b128); threads 128-191 stage B
//    (4 K-rows x 8 cols), repacking so the transposed store is 8x ds_store_b64.
// ---------------------------------------------------------------------------
#define AST 40   // LDS row pitch (halves) for A tile (80B, 16B aligned)
#define BST 40   // LDS row pitch (halves) for transposed B tile
__global__ __launch_bounds__(256)
void k_gemm_f16(const _Float16* __restrict__ A, const _Float16* __restrict__ Wt,
                const float* __restrict__ bias, const float* __restrict__ res,
                float* __restrict__ outF, _Float16* __restrict__ outH,
                int M, int N, int K, int act) {
    __shared__ _Float16 sA[2][128 * AST];   // [128 rows][32 halves + pad]
    __shared__ _Float16 sB[2][64 * BST];    // transposed: [64 cols][32 halves + pad]
    const int tid = threadIdx.x;
    const int wave = tid >> 5, lane = tid & 31;
    const int mbase = blockIdx.y * 128, nbase = blockIdx.x * 64;

    const v8f vz = {0.f, 0.f, 0.f, 0.f, 0.f, 0.f, 0.f, 0.f};
    v8f acc[4];
    acc[0] = vz; acc[1] = vz; acc[2] = vz; acc[3] = vz;

    // A-layout: lane<16 -> K 0..15; lane>=16 -> K 16..31 (matches B)
    const int ar  = wave * 16 + (lane & 15);
    const int kbA = (lane >= 16) ? 16 : 0;
    // B-layout (contiguous K, same as A): lane<16 -> K 0..15; lane>=16 -> K 16..31; col = lane&15 (+16f)
    const int kbB = (lane >= 16) ? 16 : 0;

    // loader roles
    const bool isA = (tid < 128);
    const bool isB = (tid >= 128 && tid < 192);
    const int  aRow = tid;                 // 0..127 (A role)
    const int  bT   = tid - 128;           // 0..63  (B role)
    const int  bCh  = bT & 7;              // column chunk of 8
    const int  bKq  = bT >> 3;             // K group of 4 rows: rows bKq*4 .. +4

    uint4 pf[4];  // staged global data (A: 4x8 halves of one row; B: 4 K-rows x 8 cols)

    auto loadTile = [&](int k0) {
        if (isA) {
            const int grow = mbase + aRow;
            if (grow < M) {
                const uint4* src = (const uint4*)(A + (size_t)grow * K + k0);
#pragma unroll
                for (int i = 0; i < 4; ++i) pf[i] = src[i];
            } else {
                const uint4 z = make_uint4(0u, 0u, 0u, 0u);
#pragma unroll
                for (int i = 0; i < 4; ++i) pf[i] = z;
            }
        } else if (isB) {
#pragma unroll
            for (int i = 0; i < 4; ++i)
                pf[i] = *(const uint4*)(Wt + (size_t)(k0 + bKq * 4 + i) * N + nbase + bCh * 8);
        }
    };

    auto storeTile = [&](int buf) {
        if (isA) {
            uint4* dst = (uint4*)&sA[buf][aRow * AST];
#pragma unroll
            for (int i = 0; i < 4; ++i) dst[i] = pf[i];
        } else if (isB) {
            const _Float16* h0 = (const _Float16*)&pf[0];
            const _Float16* h1 = (const _Float16*)&pf[1];
            const _Float16* h2 = (const _Float16*)&pf[2];
            const _Float16* h3 = (const _Float16*)&pf[3];
#pragma unroll
            for (int i = 0; i < 8; ++i) {   // per column: 4 halves contiguous along K
                union { _Float16 h[4]; unsigned long long u; } pk;
                pk.h[0] = h0[i]; pk.h[1] = h1[i]; pk.h[2] = h2[i]; pk.h[3] = h3[i];
                *(unsigned long long*)&sB[buf][(bCh * 8 + i) * BST + bKq * 4] = pk.u;
            }
        }
    };

    loadTile(0);
    int buf = 0;
    for (int k0 = 0; k0 < K; k0 += 32) {
        storeTile(buf);
        __syncthreads();
        if (k0 + 32 < K) loadTile(k0 + 32);   // overlap next tile with WMMA below

        v16h av;
        {
            uint4* aq = (uint4*)&av;
            aq[0] = *(const uint4*)&sA[buf][ar * AST + kbA];
            aq[1] = *(const uint4*)&sA[buf][ar * AST + kbA + 8];
        }
        v16h bv[4];
#pragma unroll
        for (int f = 0; f < 4; ++f) {         // preload all B frags, then burst WMMAs
            const int bn = f * 16 + (lane & 15);
            uint4* bq = (uint4*)&bv[f];
            bq[0] = *(const uint4*)&sB[buf][bn * BST + kbB];
            bq[1] = *(const uint4*)&sB[buf][bn * BST + kbB + 8];
        }
#pragma unroll
        for (int f = 0; f < 4; ++f)
            acc[f] = __builtin_amdgcn_wmma_f32_16x16x32_f16(
                false, av, false, bv[f], (short)0, acc[f], false, false);
        buf ^= 1;
    }

    // Epilogue: C layout lane<16 -> M=r, lane>=16 -> M=r+8; N = lane&15
    const int rOff = (lane < 16) ? 0 : 8;
    const int cl = lane & 15;
#pragma unroll
    for (int f = 0; f < 4; ++f) {
        const int gcol = nbase + f * 16 + cl;
        const float bvv = bias ? bias[gcol] : 0.0f;
#pragma unroll
        for (int rr = 0; rr < 8; ++rr) {
            const int grow = mbase + wave * 16 + rOff + rr;
            if (grow < M) {
                float v = acc[f][rr] + bvv;
                if (act == 1) v = gelu_f(v);
                else if (act == 2) v = fmaxf(v, 0.0f);
                if (res)  v += res[(size_t)grow * N + gcol];
                if (outF) outF[(size_t)grow * N + gcol] = v;
                if (outH) outH[(size_t)grow * N + gcol] = (_Float16)v;
            }
        }
    }
}

// ---------------------------------------------------------------------------
// 7. Attention: one wave per (batch, head, query row). DH == 32 == wave width.
//    Causal (keys 0..qi) except row 0 attends all S keys. Two-pass softmax in LDS.
// ---------------------------------------------------------------------------
__global__ __launch_bounds__(256)
void k_attention(const _Float16* __restrict__ qkv, _Float16* __restrict__ o) {
    __shared__ float sc[8][264];
    __shared__ float qs[8][32];
    const int tid = threadIdx.x;
    const int ws = tid >> 5, lane = tid & 31;
    const int wid = blockIdx.x * 8 + ws;              // 0 .. NB*HEADS*SQ-1
    const int b = wid / (HEADS * SQ);
    int rem = wid - b * HEADS * SQ;
    const int hh = rem / SQ;
    const int qi = rem - hh * SQ;

    const size_t rowQ = (size_t)(b * SQ + qi);
    qs[ws][lane] = (float)qkv[rowQ * 768 + hh * DH + lane] * 0.17677669529663687f; // 1/sqrt(32)
    __syncthreads();

    const int cnt = (qi == 0) ? SQ : (qi + 1);
    const size_t kbase = (size_t)(b * SQ) * 768 + 256 + hh * DH;

    float mloc = -1e30f;
    for (int j = lane; j < cnt; j += 32) {
        const _Float16* kr = qkv + kbase + (size_t)j * 768;
        __builtin_prefetch(kr + 32 * 768, 0, 1);      // next chunk's K row -> global_prefetch_b8
        float s = 0.0f;
#pragma unroll
        for (int d = 0; d < DH; ++d) s += qs[ws][d] * (float)kr[d];
        sc[ws][j] = s;
        mloc = fmaxf(mloc, s);
    }
#pragma unroll
    for (int m = 16; m > 0; m >>= 1) mloc = fmaxf(mloc, __shfl_xor(mloc, m, 32));

    float lsum = 0.0f;
    for (int j = lane; j < cnt; j += 32) {
        const float pe = expf(sc[ws][j] - mloc);
        sc[ws][j] = pe;
        lsum += pe;
    }
#pragma unroll
    for (int m = 16; m > 0; m >>= 1) lsum += __shfl_xor(lsum, m, 32);
    __syncthreads();

    const float inv = 1.0f / lsum;
    const size_t vbase = (size_t)(b * SQ) * 768 + 512 + hh * DH + lane;  // lane = d
    float accv = 0.0f;
    for (int j = 0; j < cnt; ++j) accv += sc[ws][j] * (float)qkv[vbase + (size_t)j * 768];
    o[rowQ * WDIM + hh * DH + lane] = (_Float16)(accv * inv);
}

// ---------------------------------------------------------------------------
// 8. cls extraction + ln_post; decoder dense layers with ReLU
// ---------------------------------------------------------------------------
__global__ void k_cls_ln(const float* __restrict__ xd, const float* __restrict__ s,
                         const float* __restrict__ b, float* __restrict__ cls) {
    __shared__ float red[256];
    const int bb = blockIdx.x, t = threadIdx.x;
    const float v = xd[(size_t)(bb * SQ) * WDIM + t];
    red[t] = v; __syncthreads();
    for (int st = 128; st > 0; st >>= 1) { if (t < st) red[t] += red[t + st]; __syncthreads(); }
    const float mean = red[0] * (1.0f / 256.0f); __syncthreads();
    const float d = v - mean;
    red[t] = d * d; __syncthreads();
    for (int st = 128; st > 0; st >>= 1) { if (t < st) red[t] += red[t + st]; __syncthreads(); }
    const float var = red[0] * (1.0f / 256.0f);
    cls[bb * WDIM + t] = d * rsqrtf(var + 1e-5f) * s[t] + b[t];
}

__global__ void k_dense_relu(const float* __restrict__ in, const float* __restrict__ w,
                             const float* __restrict__ bias, float* __restrict__ out, int K) {
    const int r = blockIdx.x, c = threadIdx.x, Ncol = blockDim.x;
    float acc = bias[c];
    for (int k = 0; k < K; ++k) acc += in[r * K + k] * w[k * Ncol + c];
    out[r * Ncol + c] = fmaxf(acc, 0.0f);
}

// ---------------------------------------------------------------------------
extern "C" void kernel_launch(void* const* d_in, const int* in_sizes, int n_in,
                              void* d_out, int out_size, void* d_ws, size_t ws_size,
                              hipStream_t stream) {
    (void)in_sizes; (void)n_in; (void)out_size; (void)ws_size;
    const float* x          = (const float*)d_in[0];
    const float* p          = (const float*)d_in[1];
    const int*   edge       = (const int*)d_in[2];
    /* d_in[3] = batch: arange(N)//256, structure exploited directly */
    const float* gcn_w      = (const float*)d_in[4];
    const float* gcn_b      = (const float*)d_in[5];
    const float* cls_tok    = (const float*)d_in[6];
    const float* ln_pre_s   = (const float*)d_in[7];
    const float* ln_pre_b   = (const float*)d_in[8];
    const float* norm1_s    = (const float*)d_in[9];
    const float* norm1_b    = (const float*)d_in[10];
    const float* in_proj_w  = (const float*)d_in[11];
    const float* in_proj_b  = (const float*)d_in[12];
    const float* out_proj_w = (const float*)d_in[13];
    const float* out_proj_b = (const float*)d_in[14];
    const float* norm2_s    = (const float*)d_in[15];
    const float* norm2_b    = (const float*)d_in[16];
    const float* ff_w1      = (const float*)d_in[17];
    const float* ff_b1      = (const float*)d_in[18];
    const float* ff_w2      = (const float*)d_in[19];
    const float* ff_b2      = (const float*)d_in[20];
    const float* ln_post_s  = (const float*)d_in[21];
    const float* ln_post_b  = (const float*)d_in[22];
    const float* dec_w1     = (const float*)d_in[23];
    const float* dec_b1     = (const float*)d_in[24];
    const float* dec_w2     = (const float*)d_in[25];
    const float* dec_b2     = (const float*)d_in[26];
    const float* dec_w3     = (const float*)d_in[27];
    const float* dec_b3     = (const float*)d_in[28];

    char* ws = (char*)d_ws;
    size_t off = 0;
    auto alloc = [&](size_t bytes) -> char* {
        char* ptr = ws + off; off = (off + bytes + 255) & ~(size_t)255; return ptr;
    };
    float*    xw    = (float*)   alloc((size_t)N_NODES * WDIM * 4);
    float*    hbuf  = (float*)   alloc((size_t)N_NODES * WDIM * 4);
    float*    deg   = (float*)   alloc((size_t)N_NODES * 4);
    float*    dinv  = (float*)   alloc((size_t)N_NODES * 4);
    float*    xd    = (float*)   alloc((size_t)ROWS * WDIM * 4);
    _Float16* lnb   = (_Float16*)alloc((size_t)ROWS * WDIM * 2);
    _Float16* qkvh  = (_Float16*)alloc((size_t)ROWS * 768 * 2);
    _Float16* atth  = (_Float16*)alloc((size_t)ROWS * WDIM * 2);
    _Float16* ff1h  = (_Float16*)alloc((size_t)ROWS * DFFD * 2);
    _Float16* w_qkv = (_Float16*)alloc((size_t)2 * WDIM * 768 * 2);
    _Float16* w_out = (_Float16*)alloc((size_t)2 * WDIM * WDIM * 2);
    _Float16* w_f1  = (_Float16*)alloc((size_t)2 * WDIM * DFFD * 2);
    _Float16* w_f2  = (_Float16*)alloc((size_t)2 * DFFD * WDIM * 2);
    float*    clsf  = (float*)   alloc((size_t)NB * WDIM * 4);
    float*    dh1   = (float*)   alloc((size_t)NB * HIDD * 4);
    float*    dh2   = (float*)   alloc((size_t)NB * HIDD * 4);

    // Weight conversion f32 -> f16 (once per call; weights total ~3 MB)
    k_f32_f16<<<(2 * WDIM * 768 + 255) / 256, 256, 0, stream>>>(in_proj_w, w_qkv, 2 * WDIM * 768);
    k_f32_f16<<<(2 * WDIM * WDIM + 255) / 256, 256, 0, stream>>>(out_proj_w, w_out, 2 * WDIM * WDIM);
    k_f32_f16<<<(2 * WDIM * DFFD + 255) / 256, 256, 0, stream>>>(ff_w1, w_f1, 2 * WDIM * DFFD);
    k_f32_f16<<<(2 * DFFD * WDIM + 255) / 256, 256, 0, stream>>>(ff_w2, w_f2, 2 * DFFD * WDIM);

    // GCN
    k_pe_gcn_xw<<<N_NODES, 256, 0, stream>>>(x, p, gcn_w, xw);
    k_fill_deg<<<N_NODES / 256, 256, 0, stream>>>(deg);
    k_deg_edges<<<NE / 256, 256, 0, stream>>>(edge, deg);
    k_dinv<<<N_NODES / 256, 256, 0, stream>>>(deg, dinv);
    k_h_init<<<N_NODES, 256, 0, stream>>>(xw, dinv, gcn_b, hbuf);
    k_h_edges<<<(NE * 64) / 256, 256, 0, stream>>>(edge, dinv, xw, hbuf);

    // Tokens + ln_pre
    k_build_xd<<<ROWS, 256, 0, stream>>>(hbuf, cls_tok, ln_pre_s, ln_pre_b, xd);

    const int MGRID = (ROWS + 127) / 128;
    for (int l = 0; l < 2; ++l) {
        k_layernorm_f16<<<ROWS, 256, 0, stream>>>(xd, norm1_s + l * WDIM, norm1_b + l * WDIM, lnb);
        k_gemm_f16<<<dim3(768 / 64, MGRID), 256, 0, stream>>>(
            lnb, w_qkv + (size_t)l * WDIM * 768, in_proj_b + l * 768,
            nullptr, nullptr, qkvh, ROWS, 768, WDIM, 0);
        k_attention<<<ROWS, 256, 0, stream>>>(qkvh, atth);   // NB*HEADS*SQ = 8*ROWS waves
        k_gemm_f16<<<dim3(WDIM / 64, MGRID), 256, 0, stream>>>(
            atth, w_out + (size_t)l * WDIM * WDIM, out_proj_b + l * WDIM,
            xd, xd, nullptr, ROWS, WDIM, WDIM, 0);
        k_layernorm_f16<<<ROWS, 256, 0, stream>>>(xd, norm2_s + l * WDIM, norm2_b + l * WDIM, lnb);
        k_gemm_f16<<<dim3(DFFD / 64, MGRID), 256, 0, stream>>>(
            lnb, w_f1 + (size_t)l * WDIM * DFFD, ff_b1 + l * DFFD,
            nullptr, nullptr, ff1h, ROWS, DFFD, WDIM, 1);
        k_gemm_f16<<<dim3(WDIM / 64, MGRID), 256, 0, stream>>>(
            ff1h, w_f2 + (size_t)l * DFFD * WDIM, ff_b2 + l * WDIM,
            xd, xd, nullptr, ROWS, WDIM, DFFD, 1);
    }

    // cls -> ln_post -> decoder (ReLU after every layer)
    k_cls_ln<<<NB, 256, 0, stream>>>(xd, ln_post_s, ln_post_b, clsf);
    k_dense_relu<<<NB, HIDD, 0, stream>>>(clsf, dec_w1, dec_b1, dh1, WDIM);
    k_dense_relu<<<NB, HIDD, 0, stream>>>(dh1, dec_w2, dec_b2, dh2, HIDD);
    k_dense_relu<<<NB, OUTD, 0, stream>>>(dh2, dec_w3, dec_b3, (float*)d_out, HIDD);
}
